// Attention_pps_1262720385625
// MI455X (gfx1250) — matrix operand order repaired and checked by reading, not compiled, not run on hardware
//
#include <hip/hip_runtime.h>

// ---------------------------------------------------------------------------
// Problem constants (from reference): B=65536, C=512, H=8, n=1 per stream.
// Softmax over a length-1 axis == 1, so the whole op reduces to
//   out = x[:,0,:] @ (W_kv[:,C:2C] @ W_proj) + b_proj
// ---------------------------------------------------------------------------
#define BATCH 65536
#define CDIM  512

typedef __attribute__((ext_vector_type(16))) __bf16 bf16x16;
typedef __attribute__((ext_vector_type(8)))  __bf16 bf16x8;
typedef __attribute__((ext_vector_type(4)))  __bf16 bf16x4;
typedef __attribute__((ext_vector_type(8)))  float  v8f;

// ---------------------------------------------------------------------------
// Kernel 1: W_fused = W_v @ W_proj  (512x512x512, trivial), written directly
// in the WMMA B-operand fragment layout as a compensated bf16 hi/lo pair.
// Fragment = 32(K) x 16(N) bf16 tile; lane l holds column (l&15),
// K half (l&16 ? 16..31 : 0..15), 16 bf16 (32 bytes) contiguous per lane.
// frag id = nt*16 + kt  (K-tiles contiguous for the streaming k-loop).
// ---------------------------------------------------------------------------
__global__ __launch_bounds__(256) void fuse_weights_kernel(
    const float* __restrict__ Wkv, const float* __restrict__ Wp,
    __bf16* __restrict__ Bhi, __bf16* __restrict__ Blo) {
  const int t = blockIdx.x * 256 + threadIdx.x;  // 0 .. 512*512-1
  const int r = t >> 9;   // K index of W_fused
  const int n = t & 511;  // N index
  const float* a = Wkv + (size_t)r * (2 * CDIM) + CDIM;  // W_v row r
  const float* b = Wp + n;                               // W_proj col n
  float acc = 0.f;
#pragma unroll 8
  for (int c = 0; c < CDIM; ++c) acc += a[c] * b[(size_t)c * CDIM];

  const __bf16 hi = (__bf16)acc;
  const __bf16 lo = (__bf16)(acc - (float)hi);

  const int kt = r >> 5, kr = r & 31, nt = n >> 4, nl = n & 15;
  const int frag = nt * 16 + kt;
  const int lane = nl + (kr & 16);
  const int pos  = kr & 15;                       // element within lane chunk
  const int idx  = frag * 512 + lane * 16 + pos;  // bf16 elements
  Bhi[idx] = hi;
  Blo[idx] = lo;
}

// ---------------------------------------------------------------------------
// Kernel 2: out[m, :] = x_img[m, :] @ W_fused + bias
// Block = 256 threads (8 wave32), tile 64(M) x 512(N), K=512.
// Each wave: 16 rows x 256 cols = 16 accumulator fragments.
// Compensated bf16: acc += Ahi*Bhi + Ahi*Blo + Alo*Bhi  (3 WMMAs / fragment).
// ---------------------------------------------------------------------------
#define LDS_STRIDE 520  // 512 + 8 bf16 pad: 1040 B rows, 16B aligned, no bank conflicts

__global__ __launch_bounds__(256) void fused_gemm_kernel(
    const float* __restrict__ X,        // (B, 2, 512); img token at row*1024
    const __bf16* __restrict__ Bhi,
    const __bf16* __restrict__ Blo,
    const float* __restrict__ bias,
    float* __restrict__ Out) {
  __shared__ __align__(16) __bf16 Ahi[64 * LDS_STRIDE];
  __shared__ __align__(16) __bf16 Alo[64 * LDS_STRIDE];

  const int tid = threadIdx.x;
  const int m0  = blockIdx.x * 64;

  // ---- Stage A tile: 64 rows x 512 fp32 -> bf16 hi/lo split in LDS ----
#pragma unroll
  for (int i = 0; i < 32; ++i) {
    const int idx = i * 256 + tid;  // float4 index, 0..8191
    const int row = idx >> 7;       // 128 float4 per row
    const int c4  = idx & 127;
    const float4 v =
        ((const float4*)(X + (size_t)(m0 + row) * (2 * CDIM)))[c4];
    const __bf16 h0 = (__bf16)v.x, h1 = (__bf16)v.y,
                 h2 = (__bf16)v.z, h3 = (__bf16)v.w;
    const __bf16 l0 = (__bf16)(v.x - (float)h0), l1 = (__bf16)(v.y - (float)h1),
                 l2 = (__bf16)(v.z - (float)h2), l3 = (__bf16)(v.w - (float)h3);
    *(bf16x4*)&Ahi[row * LDS_STRIDE + c4 * 4] = (bf16x4){h0, h1, h2, h3};
    *(bf16x4*)&Alo[row * LDS_STRIDE + c4 * 4] = (bf16x4){l0, l1, l2, l3};
  }
  __syncthreads();

  const int wave = tid >> 5;
  const int lane = tid & 31;
  const int mw = wave & 3;   // m-subtile (16 rows each)
  const int nw = wave >> 2;  // n-group (256 cols each)

  // A-operand lane addressing (contiguous k order, same as the B packing):
  // lane<16 : K {0..15};  lane>=16 : K {16..31}
  const int rowA = mw * 16 + (lane & 15);
  const int kofs = (lane & 16) ? 16 : 0;

  v8f acc[16];
  const v8f vzero = {0.f, 0.f, 0.f, 0.f, 0.f, 0.f, 0.f, 0.f};
#pragma unroll
  for (int i = 0; i < 16; ++i) acc[i] = vzero;

  for (int kt = 0; kt < 16; ++kt) {
    const __bf16* ah = &Ahi[rowA * LDS_STRIDE + kt * 32 + kofs];
    const __bf16* al = &Alo[rowA * LDS_STRIDE + kt * 32 + kofs];
    const bf16x8 ah0 = *(const bf16x8*)ah;
    const bf16x8 ah1 = *(const bf16x8*)(ah + 8);
    const bf16x8 al0 = *(const bf16x8*)al;
    const bf16x8 al1 = *(const bf16x8*)(al + 8);
    const bf16x16 a_hi = __builtin_shufflevector(
        ah0, ah1, 0, 1, 2, 3, 4, 5, 6, 7, 8, 9, 10, 11, 12, 13, 14, 15);
    const bf16x16 a_lo = __builtin_shufflevector(
        al0, al1, 0, 1, 2, 3, 4, 5, 6, 7, 8, 9, 10, 11, 12, 13, 14, 15);

#pragma unroll
    for (int nf = 0; nf < 16; ++nf) {
      const int frag = (nw * 16 + nf) * 16 + kt;
      const size_t off = (size_t)frag * 512 + lane * 16;
      const bf16x16 b_hi = *(const bf16x16*)(Bhi + off);
      const bf16x16 b_lo = *(const bf16x16*)(Blo + off);
      acc[nf] = __builtin_amdgcn_wmma_f32_16x16x32_bf16(
          false, a_hi, false, b_hi, (short)0, acc[nf], false, false);
      acc[nf] = __builtin_amdgcn_wmma_f32_16x16x32_bf16(
          false, a_hi, false, b_lo, (short)0, acc[nf], false, false);
      acc[nf] = __builtin_amdgcn_wmma_f32_16x16x32_bf16(
          false, a_lo, false, b_hi, (short)0, acc[nf], false, false);
    }
  }

  // ---- Store: C/D 16x16 f32 layout: lanes 0-15 -> M=v, lanes 16-31 -> M=v+8
  const int col   = lane & 15;
  const int rbase = m0 + mw * 16 + ((lane & 16) ? 8 : 0);
#pragma unroll
  for (int nf = 0; nf < 16; ++nf) {
    const int n0 = (nw * 16 + nf) * 16;
    const float bv = bias[n0 + col];
    float* o = Out + (size_t)rbase * CDIM + n0 + col;
#pragma unroll
    for (int v = 0; v < 8; ++v) o[(size_t)v * CDIM] = acc[nf][v] + bv;
  }
}

// ---------------------------------------------------------------------------
extern "C" void kernel_launch(void* const* d_in, const int* in_sizes, int n_in,
                              void* d_out, int out_size, void* d_ws,
                              size_t ws_size, hipStream_t stream) {
  (void)in_sizes; (void)n_in; (void)out_size; (void)ws_size;
  const float* x    = (const float*)d_in[0];  // (B, 2, C) fp32
  const float* Wkv  = (const float*)d_in[1];  // (C, 2C)
  const float* Wp   = (const float*)d_in[2];  // (C, C)
  const float* bias = (const float*)d_in[3];  // (C,)
  float* out = (float*)d_out;                 // (B, 1, C) -> (B, C)

  __bf16* Bhi = (__bf16*)d_ws;                // 512 KB
  __bf16* Blo = Bhi + (size_t)CDIM * CDIM;    // 512 KB

  fuse_weights_kernel<<<(CDIM * CDIM) / 256, 256, 0, stream>>>(Wkv, Wp, Bhi, Blo);
  fused_gemm_kernel<<<BATCH / 64, 256, 0, stream>>>(x, Bhi, Blo, bias, out);
}
